// Mamba_Diffusion_DiT_68822555951846
// MI455X (gfx1250) — matrix operand order repaired and checked by reading, not compiled, not run on hardware
//
#include <hip/hip_runtime.h>
#include <hip/hip_bf16.h>

// ---------------- CDNA5 WMMA types ----------------
typedef __attribute__((ext_vector_type(16))) _Float16 v16h;
typedef __attribute__((ext_vector_type(8)))  float    v8f;

// ---------------- problem constants ----------------
#define H_   1024
#define I_   512
#define DI_  2048
#define TDI_ 4096
#define DTR_ 64
#define DS_  16
#define DC_  4
#define B_   4
#define L_   1024
#define LC_  120
#define NH_  16
#define DH_  64
#define M_   4096   // B_*L_
#define XPN_ 96     // DTR_ + 2*DS_

// ---------------- activations ----------------
template<int ACT>
__device__ __forceinline__ float act_of(float x) {
  if (ACT == 1) return 0.5f * x * (1.f + erff(x * 0.70710678118654752f));  // exact gelu
  if (ACT == 2) return x / (1.f + __expf(-x));                            // silu
  if (ACT == 3) return (x > 20.f) ? x : log1pf(__expf(x));                // softplus
  return x;
}

// ---------------- CDNA5 async global->LDS helpers ----------------
__device__ __forceinline__ uint32_t lds_off(const void* p) {
  // addrspace(3) flat address: low 32 bits are the LDS byte offset
  return (uint32_t)(uintptr_t)p;
}
__device__ __forceinline__ void async_ld_b128(uint32_t ldsByteOff, const float* g) {
  // GLOBAL_LOAD_ASYNC_TO_LDS_B128 (ASYNCcnt-tracked DMA into LDS)
  asm volatile("global_load_async_to_lds_b128 %0, %1, off"
               :: "v"(ldsByteOff), "v"(g)
               : "memory");
}
__device__ __forceinline__ void wait_async0() {
#if __has_builtin(__builtin_amdgcn_s_wait_asynccnt)
  __builtin_amdgcn_s_wait_asynccnt(0);
#else
  asm volatile("s_wait_asynccnt 0x0" ::: "memory");
#endif
}

// ======================================================================
// WMMA GEMM:  C[M,N] = epilogue( A[M,K] @ W[N,K]^T + bias )  (+ resid)
//   - A optionally read row-flipped within each length-Lseg segment
//   - block: 256 threads = 8 waves; wave tile 32x32 (2x2 WMMA tiles)
//   - block tile 64 (M) x 128 (N); K stepped by 32
//   - K-slabs staged global->LDS with async b128 DMA, double buffered
// ======================================================================
template<int ACT>
__global__ __launch_bounds__(256)
void wmma_gemm(const float* __restrict__ A, const float* __restrict__ W,
               const float* __restrict__ bias, const float* resid,
               float* C, int M, int N, int K, int lda, int ldc,
               int flipA, int Lseg)
{
  __shared__ __align__(16) float sA[2][64 * 32];    //  8 KB per buffer
  __shared__ __align__(16) float sB[2][128 * 32];   // 16 KB per buffer

  const int tid   = threadIdx.x;
  const int lane  = tid & 31;
  const int wave  = tid >> 5;
  const int wm    = wave & 1;
  const int wn    = wave >> 1;
  const int mlane = lane & 15;
  const int half  = lane >> 4;
  const int m0    = blockIdx.y * 64;
  const int n0    = blockIdx.x * 128;

  // ---- per-thread staging descriptors (row-clamped; flip applied here) ----
  const float* aSrc[2]; uint32_t aDst[2];
#pragma unroll
  for (int j = 0; j < 2; ++j) {
    const int q = tid + j * 256;       // 0..511
    const int r = q >> 3, c = q & 7;   // row 0..63, b128 chunk 0..7
    int m  = m0 + r;
    int mm = (m < M) ? m : (M - 1);
    if (flipA) { int bb = mm / Lseg, ll = mm % Lseg; mm = bb * Lseg + (Lseg - 1 - ll); }
    aSrc[j] = A + (long)mm * lda + c * 4;
    aDst[j] = lds_off(&sA[0][r * 32 + c * 4]);
  }
  const float* bSrc[4]; uint32_t bDst[4];
#pragma unroll
  for (int j = 0; j < 4; ++j) {
    const int q = tid + j * 256;       // 0..1023
    const int r = q >> 3, c = q & 7;   // row 0..127, chunk 0..7
    int n  = n0 + r;
    int nn = (n < N) ? n : (N - 1);
    bSrc[j] = W + (long)nn * K + c * 4;
    bDst[j] = lds_off(&sB[0][r * 32 + c * 4]);
  }
  const uint32_t aStride = 64 * 32 * 4;   // bytes between double buffers
  const uint32_t bStride = 128 * 32 * 4;

  auto stage = [&](int kb, int buf) {
#pragma unroll
    for (int j = 0; j < 2; ++j) async_ld_b128(aDst[j] + buf * aStride, aSrc[j] + kb);
#pragma unroll
    for (int j = 0; j < 4; ++j) async_ld_b128(bDst[j] + buf * bStride, bSrc[j] + kb);
  };

  // ---- fragment masks (edge tiles only deviate from 1.0) ----
  float amask[2], bmask[2];
#pragma unroll
  for (int tm = 0; tm < 2; ++tm)
    amask[tm] = ((m0 + wm * 32 + tm * 16 + mlane) < M) ? 1.f : 0.f;
#pragma unroll
  for (int tn = 0; tn < 2; ++tn)
    bmask[tn] = ((n0 + wn * 32 + tn * 16 + mlane) < N) ? 1.f : 0.f;

  v8f acc[2][2] = {};

  stage(0, 0);
  wait_async0();
  __syncthreads();

  int buf = 0;
  for (int kb = 0; kb < K; kb += 32) {
    if (kb + 32 < K) stage(kb + 32, buf ^ 1);

    const float* As = sA[buf];
    const float* Bs = sB[buf];

    v16h af[2], bf[2];
    // A fragment: 16x32 f16 per ISA 7.12.2 (lane M = lane%16, K-halves)
#pragma unroll
    for (int tm = 0; tm < 2; ++tm) {
      const float* ar = As + (wm * 32 + tm * 16 + mlane) * 32;
      const float msk = amask[tm];
#pragma unroll
      for (int i = 0; i < 4; ++i) {
        af[tm][2*i]         = (_Float16)(msk * ar[8*half + 2*i]);
        af[tm][2*i + 1]     = (_Float16)(msk * ar[8*half + 2*i + 1]);
        af[tm][8 + 2*i]     = (_Float16)(msk * ar[16 + 8*half + 2*i]);
        af[tm][8 + 2*i + 1] = (_Float16)(msk * ar[16 + 8*half + 2*i + 1]);
      }
    }
    // B fragment: 32x16 f16 (lane N = lane%16, K-halves matching A)
#pragma unroll
    for (int tn = 0; tn < 2; ++tn) {
      const float* br = Bs + (wn * 32 + tn * 16 + mlane) * 32 + 8 * half;
      const float msk = bmask[tn];
#pragma unroll
      for (int e = 0; e < 16; ++e) bf[tn][e] = (_Float16)(msk * br[e < 8 ? e : e + 8]);
    }
#pragma unroll
    for (int tm = 0; tm < 2; ++tm)
#pragma unroll
      for (int tn = 0; tn < 2; ++tn)
        acc[tm][tn] = __builtin_amdgcn_wmma_f32_16x16x32_f16(
            false, af[tm], false, bf[tn], (short)0, acc[tm][tn], false, false);

    wait_async0();       // next slab's DMA complete (this wave)
    __syncthreads();     // whole block done reading current slab
    buf ^= 1;
  }

  // epilogue: C/D layout — lane N = lane%16, VGPR r -> M = r + 8*(lane/16)
#pragma unroll
  for (int tm = 0; tm < 2; ++tm) {
#pragma unroll
    for (int tn = 0; tn < 2; ++tn) {
      int n = n0 + wn * 32 + tn * 16 + mlane;
      if (n >= N) continue;
      float bv = bias ? bias[n] : 0.f;
#pragma unroll
      for (int r = 0; r < 8; ++r) {
        int m = m0 + wm * 32 + tm * 16 + r + 8 * half;
        if (m >= M) continue;
        float vv = act_of<ACT>(acc[tm][tn][r] + bv);
        if (resid) vv += resid[(long)m * ldc + n];
        C[(long)m * ldc + n] = vv;
      }
    }
  }
}

// ======================================================================
// LayerNorm (no affine) + adaLN modulate:  out = ln(x)*(1+scale[b]) + shift[b]
// ======================================================================
__global__ __launch_bounds__(256)
void ln_modulate_k(const float* __restrict__ x, const float* __restrict__ mod,
                   float* __restrict__ out)
{
  const int row = blockIdx.x;           // b*L + l
  const int b   = row / L_;
  const float* xr = x + (long)row * H_;
  __shared__ float sred[256];

  float s = 0.f;
  for (int i = threadIdx.x; i < H_; i += 256) s += xr[i];
  sred[threadIdx.x] = s; __syncthreads();
  for (int st = 128; st > 0; st >>= 1) {
    if (threadIdx.x < st) sred[threadIdx.x] += sred[threadIdx.x + st];
    __syncthreads();
  }
  const float mean = sred[0] * (1.f / H_);
  __syncthreads();

  float v = 0.f;
  for (int i = threadIdx.x; i < H_; i += 256) { float d = xr[i] - mean; v += d * d; }
  sred[threadIdx.x] = v; __syncthreads();
  for (int st = 128; st > 0; st >>= 1) {
    if (threadIdx.x < st) sred[threadIdx.x] += sred[threadIdx.x + st];
    __syncthreads();
  }
  const float inv = rsqrtf(sred[0] * (1.f / H_) + 1e-6f);

  const float* shift = mod + (long)b * (3 * H_);
  const float* scale = shift + H_;
  for (int i = threadIdx.x; i < H_; i += 256)
    out[(long)row * H_ + i] = (xr[i] - mean) * inv * (1.f + scale[i]) + shift[i];
}

// LayerNorm with affine weight/bias
__global__ __launch_bounds__(256)
void ln_affine_k(const float* __restrict__ x, const float* __restrict__ w,
                 const float* __restrict__ bb, float* __restrict__ out)
{
  const int row = blockIdx.x;
  const float* xr = x + (long)row * H_;
  __shared__ float sred[256];

  float s = 0.f;
  for (int i = threadIdx.x; i < H_; i += 256) s += xr[i];
  sred[threadIdx.x] = s; __syncthreads();
  for (int st = 128; st > 0; st >>= 1) {
    if (threadIdx.x < st) sred[threadIdx.x] += sred[threadIdx.x + st];
    __syncthreads();
  }
  const float mean = sred[0] * (1.f / H_);
  __syncthreads();

  float v = 0.f;
  for (int i = threadIdx.x; i < H_; i += 256) { float d = xr[i] - mean; v += d * d; }
  sred[threadIdx.x] = v; __syncthreads();
  for (int st = 128; st > 0; st >>= 1) {
    if (threadIdx.x < st) sred[threadIdx.x] += sred[threadIdx.x + st];
    __syncthreads();
  }
  const float inv = rsqrtf(sred[0] * (1.f / H_) + 1e-6f);

  for (int i = threadIdx.x; i < H_; i += 256)
    out[(long)row * H_ + i] = (xr[i] - mean) * inv * w[i] + bb[i];
}

// ======================================================================
// Depthwise causal conv (DC=4) over L + SiLU.  Input = xm view of xz.
// ======================================================================
__global__ __launch_bounds__(256)
void dwconv_silu_k(const float* __restrict__ xz, const float* __restrict__ cw,
                   const float* __restrict__ cb, float* __restrict__ xc)
{
  long idx = (long)blockIdx.x * 256 + threadIdx.x;
  if (idx >= (long)M_ * DI_) return;
  const int  d    = (int)(idx % DI_);
  const long row  = idx / DI_;
  const int  l    = (int)(row % L_);
  const long brow = row - l;
  float acc = cb[d];
#pragma unroll
  for (int j = 0; j < DC_; ++j) {
    int ls = l - (DC_ - 1) + j;
    if (ls >= 0) acc += cw[d * DC_ + j] * xz[(brow + ls) * TDI_ + d];
  }
  xc[row * DI_ + d] = acc / (1.f + __expf(-acc));
}

// ======================================================================
// Selective scan: one thread per (b,d) strand, 16-state recurrence in regs.
// Writes y in-place over xcy (= silu'd conv output), fused with z-gating.
// ======================================================================
__global__ __launch_bounds__(256)
void scan_k(const float* __restrict__ dt, float* xcy,
            const float* __restrict__ xdbl, const float* __restrict__ xz,
            const float* __restrict__ A_log, const float* __restrict__ Dp)
{
  const int d = blockIdx.x * 256 + threadIdx.x;   // 0..DI_-1
  const int b = blockIdx.y;

  float Arow[DS_];
#pragma unroll
  for (int n = 0; n < DS_; ++n) Arow[n] = -__expf(A_log[d * DS_ + n]);
  const float Dd = Dp[d];

  float h[DS_];
#pragma unroll
  for (int n = 0; n < DS_; ++n) h[n] = 0.f;

  __shared__ float sB[DS_], sC[DS_];
  const long base = (long)b * L_;

  for (int l = 0; l < L_; ++l) {
    const long row = base + l;
    if (threadIdx.x < DS_)
      sB[threadIdx.x] = xdbl[row * XPN_ + DTR_ + threadIdx.x];
    else if (threadIdx.x < 2 * DS_)
      sC[threadIdx.x - DS_] = xdbl[row * XPN_ + DTR_ + DS_ + (threadIdx.x - DS_)];
    __syncthreads();

    const float dtv = dt[row * DI_ + d];
    const float xv  = xcy[row * DI_ + d];
    const float dtx = dtv * xv;
    float accv = 0.f;
#pragma unroll
    for (int n = 0; n < DS_; ++n) {
      h[n] = __expf(dtv * Arow[n]) * h[n] + dtx * sB[n];
      accv += h[n] * sC[n];
    }
    const float zv = xz[row * TDI_ + DI_ + d];
    xcy[row * DI_ + d] = (accv + xv * Dd) * (zv / (1.f + __expf(-zv)));
    __syncthreads();
  }
}

// ======================================================================
// Cross-attention (Dh=64, LC=120): online-softmax, one thread per query row.
// ======================================================================
__global__ __launch_bounds__(128)
void cross_attn_k(const float* __restrict__ q, const float* __restrict__ kv,
                  float* __restrict__ o)
{
  const int qi = blockIdx.x * 128 + threadIdx.x;
  const int h  = blockIdx.y;
  const int b  = blockIdx.z;
  if (qi >= L_) return;

  const float* qp = q + ((long)(b * L_ + qi)) * H_ + h * DH_;
  float qv[DH_];
#pragma unroll
  for (int i = 0; i < DH_; ++i) qv[i] = qp[i] * 0.125f;  // 1/sqrt(64)

  float mrun = -1e30f, ssum = 0.f, ov[DH_];
#pragma unroll
  for (int i = 0; i < DH_; ++i) ov[i] = 0.f;

  for (int k = 0; k < LC_; ++k) {
    const float* kp = kv + ((long)(b * LC_ + k)) * (2 * H_) + h * DH_;
    float s = 0.f;
#pragma unroll
    for (int i = 0; i < DH_; ++i) s += qv[i] * kp[i];
    const float mn   = fmaxf(mrun, s);
    const float corr = __expf(mrun - mn);
    const float w    = __expf(s - mn);
    ssum = ssum * corr + w;
    const float* vp = kp + H_;
#pragma unroll
    for (int i = 0; i < DH_; ++i) ov[i] = ov[i] * corr + w * vp[i];
    mrun = mn;
  }
  const float inv = 1.f / ssum;
  float* op = o + ((long)(b * L_ + qi)) * H_ + h * DH_;
#pragma unroll
  for (int i = 0; i < DH_; ++i) op[i] = ov[i] * inv;
}

// ---------------- small elementwise kernels ----------------
__global__ void silu_vec_k(const float* __restrict__ a, float* __restrict__ o, long n) {
  long i = (long)blockIdx.x * 256 + threadIdx.x;
  if (i < n) { float v = a[i]; o[i] = v / (1.f + __expf(-v)); }
}
__global__ void mul_inplace_k(float* a, const float* __restrict__ b, long n) {
  long i = (long)blockIdx.x * 256 + threadIdx.x;
  if (i < n) a[i] *= b[i];
}
__global__ void final_out_k(const float* __restrict__ x, const float* __restrict__ mod,
                            const float* __restrict__ fuse, float* __restrict__ out) {
  long i = (long)blockIdx.x * 256 + threadIdx.x;
  if (i >= (long)M_ * H_) return;
  const int  c   = (int)(i % H_);
  const long row = i / H_;
  const int  b   = (int)(row / L_);
  out[i] = x[i] + mod[(long)b * (3 * H_) + 2 * H_ + c] * fuse[i];
}

// ---------------- host-side helpers ----------------
static inline void launch_gemm(int act, const float* A, const float* W, const float* bias,
                               const float* resid, float* C, int M, int N, int K,
                               int lda, int ldc, int flipA, hipStream_t s)
{
  dim3 grid((N + 127) / 128, (M + 63) / 64);
  dim3 block(256);
  switch (act) {
    case 1: wmma_gemm<1><<<grid, block, 0, s>>>(A, W, bias, resid, C, M, N, K, lda, ldc, flipA, L_); break;
    case 2: wmma_gemm<2><<<grid, block, 0, s>>>(A, W, bias, resid, C, M, N, K, lda, ldc, flipA, L_); break;
    case 3: wmma_gemm<3><<<grid, block, 0, s>>>(A, W, bias, resid, C, M, N, K, lda, ldc, flipA, L_); break;
    default: wmma_gemm<0><<<grid, block, 0, s>>>(A, W, bias, resid, C, M, N, K, lda, ldc, flipA, L_); break;
  }
}

struct MambaP {
  const float *in_proj_w, *conv_w, *conv_b, *x_proj_w, *dt_proj_w, *dt_proj_b,
              *A_log, *D, *out_proj_w;
};

static void run_mamba(const float* u, const MambaP& p,
                      float* xz, float* xc, float* dtb, float* xdbl,
                      float* out, hipStream_t s)
{
  // in_proj: [M,1024] x [4096,1024]^T -> xz [M,4096]
  launch_gemm(0, u, p.in_proj_w, nullptr, nullptr, xz, M_, TDI_, H_, H_, TDI_, 0, s);
  // depthwise causal conv + silu -> xc [M,2048]
  {
    long total = (long)M_ * DI_;
    dwconv_silu_k<<<(unsigned)((total + 255) / 256), 256, 0, s>>>(xz, p.conv_w, p.conv_b, xc);
  }
  // x_proj: [M,2048] x [96,2048]^T -> xdbl [M,96]
  launch_gemm(0, xc, p.x_proj_w, nullptr, nullptr, xdbl, M_, XPN_, DI_, DI_, XPN_, 0, s);
  // dt_proj + softplus: [M,64] (view of xdbl, lda=96) x [2048,64]^T -> dtb [M,2048]
  launch_gemm(3, xdbl, p.dt_proj_w, p.dt_proj_b, nullptr, dtb, M_, DI_, DTR_, XPN_, DI_, 0, s);
  // selective scan (+ D skip + z gating), writes y in-place over xc
  scan_k<<<dim3(DI_ / 256, B_), 256, 0, s>>>(dtb, xc, xdbl, xz, p.A_log, p.D);
  // out_proj: [M,2048] x [1024,2048]^T -> out [M,1024]
  launch_gemm(0, xc, p.out_proj_w, nullptr, nullptr, out, M_, H_, DI_, DI_, H_, 0, s);
}

// ======================================================================
extern "C" void kernel_launch(void* const* d_in, const int* in_sizes, int n_in,
                              void* d_out, int out_size, void* d_ws, size_t ws_size,
                              hipStream_t stream)
{
  (void)in_sizes; (void)n_in; (void)out_size; (void)ws_size;
  int p = 0;
  const float* x       = (const float*)d_in[p++];
  const float* t       = (const float*)d_in[p++];
  const float* context = (const float*)d_in[p++];
  const float* ada_w   = (const float*)d_in[p++]; const float* ada_b   = (const float*)d_in[p++];
  const float* hgd1_w  = (const float*)d_in[p++]; const float* hgd1_b  = (const float*)d_in[p++];
  const float* hgd2_w  = (const float*)d_in[p++]; const float* hgd2_b  = (const float*)d_in[p++];
  const float* ln_w    = (const float*)d_in[p++]; const float* ln_b    = (const float*)d_in[p++];
  const float* dv_w    = (const float*)d_in[p++]; const float* dv_b    = (const float*)d_in[p++];
  const float* duf_w   = (const float*)d_in[p++]; const float* duf_b   = (const float*)d_in[p++];
  const float* dub_w   = (const float*)d_in[p++]; const float* dub_b   = (const float*)d_in[p++];
  const float* ducf_w  = (const float*)d_in[p++]; const float* ducf_b  = (const float*)d_in[p++];
  const float* ducb_w  = (const float*)d_in[p++]; const float* ducb_b  = (const float*)d_in[p++];
  const float* dol_w   = (const float*)d_in[p++]; const float* dol_b   = (const float*)d_in[p++];
  const float* do_w    = (const float*)d_in[p++]; const float* do_b    = (const float*)d_in[p++];
  MambaP mf, mb;
  mf.in_proj_w = (const float*)d_in[p++]; mf.conv_w   = (const float*)d_in[p++];
  mf.conv_b    = (const float*)d_in[p++]; mf.x_proj_w = (const float*)d_in[p++];
  mf.dt_proj_w = (const float*)d_in[p++]; mf.dt_proj_b= (const float*)d_in[p++];
  mf.A_log     = (const float*)d_in[p++]; mf.D        = (const float*)d_in[p++];
  mf.out_proj_w= (const float*)d_in[p++];
  mb.in_proj_w = (const float*)d_in[p++]; mb.conv_w   = (const float*)d_in[p++];
  mb.conv_b    = (const float*)d_in[p++]; mb.x_proj_w = (const float*)d_in[p++];
  mb.dt_proj_w = (const float*)d_in[p++]; mb.dt_proj_b= (const float*)d_in[p++];
  mb.A_log     = (const float*)d_in[p++]; mb.D        = (const float*)d_in[p++];
  mb.out_proj_w= (const float*)d_in[p++];
  const float* q_w     = (const float*)d_in[p++]; const float* q_b     = (const float*)d_in[p++];
  const float* kv_w    = (const float*)d_in[p++]; const float* kv_b    = (const float*)d_in[p++];
  const float* proj_w  = (const float*)d_in[p++]; const float* proj_b  = (const float*)d_in[p++];
  const float* hgfm_w  = (const float*)d_in[p++]; const float* hgfm_b  = (const float*)d_in[p++];
  const float* hgfr_w  = (const float*)d_in[p++]; const float* hgfr_b  = (const float*)d_in[p++];
  const float* hgff_w  = (const float*)d_in[p++]; const float* hgff_b  = (const float*)d_in[p++];
  float* out = (float*)d_out;

  // ---------------- workspace arena ----------------
  float* ws = (float*)d_ws;
  size_t off = 0;
  auto alloc = [&](size_t n) { float* q_ = ws + off; off += n; return q_; };
  const size_t MH = (size_t)M_ * H_, MI = (size_t)M_ * I_;
  float* tsilu = alloc(B_ * H_);
  float* mod   = alloc(B_ * 3 * H_);
  float* x1    = alloc(MH);
  float* xd    = alloc(MH);               // also the BiGS residual
  float* vbuf  = alloc(MI);
  float* fbuf  = alloc(MH);
  float* bkbuf = alloc(MH);
  float* u_f   = alloc(MH);
  float* u_b   = alloc(MH);
  float* hn    = alloc(MH);
  float* iA    = alloc(MI);
  float* xz    = alloc((size_t)M_ * TDI_); // mamba scratch (reused for attention)
  float* xc    = alloc((size_t)M_ * DI_);
  float* dtb   = alloc((size_t)M_ * DI_);
  float* xdbl  = alloc((size_t)M_ * XPN_);
  // reuse map
  float* x12  = hn;
  float* uc_f = u_f;
  float* uc_b = u_b;
  float* qb   = xz;
  float* kvb  = xz + MH;
  float* ao   = xz + 2 * MH;
  float* g    = iA;
  float* m1   = xc;
  float* m2   = xc + MI;
  float* fuse = dtb;

  // 1. mod = silu(t) @ ada_w^T + ada_b
  silu_vec_k<<<(B_ * H_ + 255) / 256, 256, 0, stream>>>(t, tsilu, (long)B_ * H_);
  launch_gemm(0, tsilu, ada_w, ada_b, nullptr, mod, B_, 3 * H_, H_, H_, 3 * H_, 0, stream);

  // 2. x1 = ln(x)*(1+scale)+shift
  ln_modulate_k<<<M_, 256, 0, stream>>>(x, mod, x1);

  // 3. xd = lin(silu(lin(x1, hgd1)), hgd2)
  launch_gemm(2, x1, hgd1_w, hgd1_b, nullptr, iA, M_, I_, H_, H_, I_, 0, stream);
  launch_gemm(0, iA, hgd2_w, hgd2_b, nullptr, xd, M_, H_, I_, I_, H_, 0, stream);

  // 4. BiGS: hn = ln_affine(xd); v/u_f/u_b = gelu projections (u_b on flipped hn)
  ln_affine_k<<<M_, 256, 0, stream>>>(xd, ln_w, ln_b, hn);
  launch_gemm(1, hn, dv_w,  dv_b,  nullptr, vbuf, M_, I_, H_, H_, I_, 0, stream);
  launch_gemm(1, hn, duf_w, duf_b, nullptr, u_f,  M_, H_, H_, H_, H_, 0, stream);
  launch_gemm(1, hn, dub_w, dub_b, nullptr, u_b,  M_, H_, H_, H_, H_, 1, stream);

  // 5. Mamba forward / backward
  run_mamba(u_f, mf, xz, xc, dtb, xdbl, fbuf,  stream);
  run_mamba(u_b, mb, xz, xc, dtb, xdbl, bkbuf, stream);

  // 6. uc_f = lin(f); uc_b = flip(lin(bk)); prod = uc_f*uc_b
  launch_gemm(0, fbuf,  ducf_w, ducf_b, nullptr, uc_f, M_, H_, H_, H_, H_, 0, stream);
  launch_gemm(0, bkbuf, ducb_w, ducb_b, nullptr, uc_b, M_, H_, H_, H_, H_, 1, stream);
  mul_inplace_k<<<(unsigned)((MH + 255) / 256), 256, 0, stream>>>(uc_f, uc_b, (long)MH);

  // 7. out_bigs = lin(gelu(lin(prod, dol)) * v, do);  x12 = xd + out_bigs
  launch_gemm(1, uc_f, dol_w, dol_b, nullptr, g, M_, I_, H_, H_, I_, 0, stream);
  mul_inplace_k<<<(unsigned)((MI + 255) / 256), 256, 0, stream>>>(g, vbuf, (long)MI);
  launch_gemm(0, g, do_w, do_b, xd, x12, M_, H_, I_, I_, H_, 0, stream);

  // 8. cross attention; x12 += proj(attn)
  launch_gemm(0, x12, q_w, q_b, nullptr, qb, M_, H_, H_, H_, H_, 0, stream);
  launch_gemm(0, context, kv_w, kv_b, nullptr, kvb, B_ * LC_, 2 * H_, H_, H_, 2 * H_, 0, stream);
  cross_attn_k<<<dim3((L_ + 127) / 128, NH_, B_), 128, 0, stream>>>(qb, kvb, ao);
  launch_gemm(0, ao, proj_w, proj_b, x12, x12, M_, H_, H_, H_, H_, 0, stream);

  // 9. fuse = lin(silu(lin(x12, main)) * silu(lin(x1, res)), hgf_fuse)
  launch_gemm(2, x12, hgfm_w, hgfm_b, nullptr, m1, M_, I_, H_, H_, I_, 0, stream);
  launch_gemm(2, x1,  hgfr_w, hgfr_b, nullptr, m2, M_, I_, H_, H_, I_, 0, stream);
  mul_inplace_k<<<(unsigned)((MI + 255) / 256), 256, 0, stream>>>(m1, m2, (long)MI);
  launch_gemm(0, m1, hgff_w, hgff_b, nullptr, fuse, M_, H_, I_, I_, H_, 0, stream);

  // 10. out = x + alpha[:,None] * fuse
  final_out_k<<<(unsigned)((MH + 255) / 256), 256, 0, stream>>>(x, mod, fuse, out);
}
